// SC_encoder_50500225466896
// MI455X (gfx1250) — matrix operand order repaired and checked by reading, not compiled, not run on hardware
//
#include <hip/hip_runtime.h>
#include <hip/hip_bf16.h>
#include <math.h>

typedef __attribute__((ext_vector_type(16))) _Float16 v16h;
typedef __attribute__((ext_vector_type(8)))  float    v8f;

#define HID 64
#define KNEI 15

// Branch-free tanh: 1 - 2/(exp(2x)+1).  Uses hardware v_exp_f32 (__expf),
// no divergent range-reduction.  exp->inf => 1, exp->0 => -1 (inf-safe).
__device__ __forceinline__ float fast_tanh(float x) {
  float t = __expf(2.0f * x);
  return 1.0f - 2.0f / (t + 1.0f);
}

// ---------------------------------------------------------------------------
// proj: out[r] = dot(h[r, 0:64], vec[0:64])   (one wave32 per row, float2/lane)
// ---------------------------------------------------------------------------
__global__ void proj_kernel(const float* __restrict__ h,
                            const float* __restrict__ vec,
                            float* __restrict__ out, int M) {
  int wid  = (blockIdx.x * blockDim.x + threadIdx.x) >> 5;
  int lane = threadIdx.x & 31;
  if (wid >= M) return;
  float2 v = *(const float2*)(h + (size_t)wid * HID + lane * 2);
  float2 a = *(const float2*)(vec + lane * 2);
  float s = v.x * a.x + v.y * a.y;
  #pragma unroll
  for (int off = 16; off; off >>= 1) s += __shfl_xor(s, off, 32);
  if (lane == 0) out[wid] = s;
}

// ---------------------------------------------------------------------------
// intra: per node, softmax over 15 scalar scores, weighted sum of gathered
// rows, ELU.  One wave32 per node; each lane owns 2 feature dims.
// score[n,k] = leaky_relu(refp[n] + neip[nei[n,k]], 0.01)
// ---------------------------------------------------------------------------
__global__ void intra_kernel(const int*   __restrict__ nei,
                             const float* __restrict__ src,
                             const float* __restrict__ refp,
                             const float* __restrict__ neip,
                             float* __restrict__ out, int N) {
  int wid  = (blockIdx.x * blockDim.x + threadIdx.x) >> 5;
  int lane = threadIdx.x & 31;
  if (wid >= N) return;

  float r = refp[wid];
  const int* nrow = nei + (size_t)wid * KNEI;

  int   idx[KNEI];
  float s[KNEI];
  float smax = -1e30f;
  #pragma unroll
  for (int k = 0; k < KNEI; ++k) {
    idx[k] = nrow[k];
    float t = r + neip[idx[k]];
    t = t > 0.0f ? t : 0.01f * t;           // leaky_relu(., 0.01)
    s[k] = t;
    smax = fmaxf(smax, t);
  }
  float ssum = 0.0f;
  #pragma unroll
  for (int k = 0; k < KNEI; ++k) { s[k] = __expf(s[k] - smax); ssum += s[k]; }
  float inv = 1.0f / ssum;

  float2 acc = {0.0f, 0.0f};
  #pragma unroll
  for (int k = 0; k < KNEI; ++k) {
    float w = s[k] * inv;
    float2 e = *(const float2*)(src + (size_t)idx[k] * HID + lane * 2);
    acc.x += w * e.x;
    acc.y += w * e.y;
  }
  // elu (alpha = 1), branch-free select
  float ex = __expf(fminf(acc.x, 0.0f)) - 1.0f;
  float ey = __expf(fminf(acc.y, 0.0f)) - 1.0f;
  acc.x = acc.x > 0.0f ? acc.x : ex;
  acc.y = acc.y > 0.0f ? acc.y : ey;
  *(float2*)(out + (size_t)wid * HID + lane * 2) = acc;
}

// ---------------------------------------------------------------------------
// fc/tanh/colsum via WMMA:  colsum[o] += sum_rows tanh( (E @ W^T)[n,o] + b[o] )
// E:[N,64] f32 (N multiple of 16), W:[64,64] row-major (W[o][i]), b:[64].
// D(16x16) = A(16x32,f16) x B(32x16,f16) + C ; K=64 -> 2 WMMAs per tile;
// 4 column tiles cover the 64 outputs.  B fragments are loop-invariant.
// ---------------------------------------------------------------------------
__global__ void __launch_bounds__(256)
fc_tanh_colsum_kernel(const float* __restrict__ E,
                      const float* __restrict__ W,
                      const float* __restrict__ bias,
                      float* __restrict__ colsum,
                      int N) {
  int lane = threadIdx.x & 31;
  int wid  = (blockIdx.x * blockDim.x + threadIdx.x) >> 5;
  int nw   = (gridDim.x * blockDim.x) >> 5;
  int c = lane & 15;   // output column within a 16-col tile / A row
  int g = lane >> 4;   // half-wave group

  // B fragments: b.h[j] = W[oc*16+c][kc*32 + g*16 + j]  (32x16 K-major layout)
  v16h bfrag[4][2];
  #pragma unroll
  for (int oc = 0; oc < 4; ++oc)
    #pragma unroll
    for (int kc = 0; kc < 2; ++kc) {
      const float* wrow = W + (size_t)(oc * 16 + c) * HID + kc * 32 + g * 16;
      v16h b;
      #pragma unroll
      for (int q = 0; q < 4; ++q) {
        float4 w4 = *(const float4*)(wrow + q * 4);
        b[q * 4 + 0] = (_Float16)w4.x;
        b[q * 4 + 1] = (_Float16)w4.y;
        b[q * 4 + 2] = (_Float16)w4.z;
        b[q * 4 + 3] = (_Float16)w4.w;
      }
      bfrag[oc][kc] = b;
    }

  float colacc[4] = {0.0f, 0.0f, 0.0f, 0.0f};
  int ntiles = N >> 4;
  for (int t = wid; t < ntiles; t += nw) {
    // A fragments: a.h[2v],a.h[2v+1] = E[t*16 + c][kbase + k], k = 16*g + 2v, contiguous
    const float* erow = E + (size_t)(t * 16 + c) * HID;
    v16h a0, a1;
    #pragma unroll
    for (int v = 0; v < 8; ++v) {
      int k = v * 2 + g * 16;  // contiguous k = 16*g + s, matches B
      float2 lo = *(const float2*)(erow + k);
      float2 hi = *(const float2*)(erow + 32 + k);
      a0[2 * v]     = (_Float16)lo.x;
      a0[2 * v + 1] = (_Float16)lo.y;
      a1[2 * v]     = (_Float16)hi.x;
      a1[2 * v + 1] = (_Float16)hi.y;
    }
    #pragma unroll
    for (int oc = 0; oc < 4; ++oc) {
      v8f cacc = {};
      cacc = __builtin_amdgcn_wmma_f32_16x16x32_f16(
          false, a0, false, bfrag[oc][0], (short)0, cacc, false, false);
      cacc = __builtin_amdgcn_wmma_f32_16x16x32_f16(
          false, a1, false, bfrag[oc][1], (short)0, cacc, false, false);
      float bv = bias[oc * 16 + c];
      float part = 0.0f;
      #pragma unroll
      for (int rr = 0; rr < 8; ++rr) part += fast_tanh(cacc[rr] + bv);
      colacc[oc] += part;   // lane holds rows {g*8 .. g*8+7} of column oc*16+c
    }
  }
  #pragma unroll
  for (int oc = 0; oc < 4; ++oc) {
    float v2 = colacc[oc] + __shfl_xor(colacc[oc], 16, 32);
    if (g == 0) atomicAdd(colsum + oc * 16 + c, v2);
  }
}

// ---------------------------------------------------------------------------
// beta: beta_c = softmax_c( sem . colsum_c / N )   (tiny, single thread)
// ---------------------------------------------------------------------------
__global__ void beta_kernel(const float* __restrict__ colsum,   // [C,64]
                            const float* __restrict__ sem,      // [64]
                            float* __restrict__ betas, int C, float invN) {
  if (threadIdx.x != 0 || blockIdx.x != 0) return;
  float b[3];
  float mx = -1e30f;
  for (int ch = 0; ch < C; ++ch) {
    float s = 0.0f;
    for (int o = 0; o < HID; ++o) s += sem[o] * colsum[ch * HID + o] * invN;
    b[ch] = s;
    mx = fmaxf(mx, s);
  }
  float sum = 0.0f;
  for (int ch = 0; ch < C; ++ch) { b[ch] = __expf(b[ch] - mx); sum += b[ch]; }
  for (int ch = 0; ch < C; ++ch) betas[ch] = b[ch] / sum;
}

// ---------------------------------------------------------------------------
// combine: z = beta0*e0 + beta1*e1 (+ beta2*e2)
// ---------------------------------------------------------------------------
__global__ void combine_kernel(const float* __restrict__ e0,
                               const float* __restrict__ e1,
                               const float* __restrict__ e2,
                               const float* __restrict__ betas, int C,
                               float* __restrict__ out, int total) {
  int i = blockIdx.x * blockDim.x + threadIdx.x;
  if (i >= total) return;
  float v = betas[0] * e0[i] + betas[1] * e1[i];
  if (C == 3) v += betas[2] * e2[i];
  out[i] = v;
}

// ---------------------------------------------------------------------------
extern "C" void kernel_launch(void* const* d_in, const int* in_sizes, int n_in,
                              void* d_out, int out_size, void* d_ws, size_t ws_size,
                              hipStream_t stream) {
  const float* h0 = (const float*)d_in[0];     // [50000,64]
  const float* h1 = (const float*)d_in[1];     // [50000,64]
  const float* h3 = (const float*)d_in[3];     // [20000,64]
  const int* nei_d0 = (const int*)d_in[4];
  const int* nei_d1 = (const int*)d_in[5];
  const int* nei_p0 = (const int*)d_in[6];
  const int* nei_p1 = (const int*)d_in[7];
  const int* nei_p2 = (const int*)d_in[8];
  const float* att_d = (const float*)d_in[9];  // [2,128]
  const float* att_p = (const float*)d_in[10]; // [3,128]
  const float* fc_w_d = (const float*)d_in[11];
  const float* fc_b_d = (const float*)d_in[12];
  const float* sem_d  = (const float*)d_in[13];
  const float* fc_w_p = (const float*)d_in[14];
  const float* fc_b_p = (const float*)d_in[15];
  const float* sem_p  = (const float*)d_in[16];

  const int N  = 50000;               // ref node count (both sides)
  const int M0 = in_sizes[0] / HID;   // 50000
  const int M1 = in_sizes[1] / HID;   // 50000
  const int M3 = in_sizes[3] / HID;   // 20000

  // workspace layout (floats)
  float* ws     = (float*)d_ws;
  float* emb    = ws;                               // 5 * N * 64
  float* refp   = emb   + (size_t)5 * N * HID;      // 5 * N
  float* neip   = refp  + (size_t)5 * N;            // 5 * N
  float* colsum = neip  + (size_t)5 * N;            // 5 * 64
  float* betas  = colsum + 5 * HID;                 // 5

  dim3 blk(256);
  auto wblocks = [](int rows) { return dim3((unsigned)((rows + 7) / 8)); };

  // --- reference-side projections: refp[m][n] = h_ref[n] . att_m[:64]
  proj_kernel<<<wblocks(N), blk, 0, stream>>>(h0, att_d + 0 * 128,       refp + 0 * N, N);
  proj_kernel<<<wblocks(N), blk, 0, stream>>>(h0, att_d + 1 * 128,       refp + 1 * N, N);
  proj_kernel<<<wblocks(N), blk, 0, stream>>>(h1, att_p + 0 * 128,       refp + 2 * N, N);
  proj_kernel<<<wblocks(N), blk, 0, stream>>>(h1, att_p + 1 * 128,       refp + 3 * N, N);
  proj_kernel<<<wblocks(N), blk, 0, stream>>>(h1, att_p + 2 * 128,       refp + 4 * N, N);
  // --- neighbor-side projections: neip[m][j] = h_src[j] . att_m[64:]
  proj_kernel<<<wblocks(M0), blk, 0, stream>>>(h0, att_d + 0 * 128 + 64, neip + 0 * N, M0);
  proj_kernel<<<wblocks(M1), blk, 0, stream>>>(h1, att_d + 1 * 128 + 64, neip + 1 * N, M1);
  proj_kernel<<<wblocks(M0), blk, 0, stream>>>(h0, att_p + 0 * 128 + 64, neip + 2 * N, M0);
  proj_kernel<<<wblocks(M1), blk, 0, stream>>>(h1, att_p + 1 * 128 + 64, neip + 3 * N, M1);
  proj_kernel<<<wblocks(M3), blk, 0, stream>>>(h3, att_p + 2 * 128 + 64, neip + 4 * N, M3);

  // --- intra attention -> elu -> embeddings
  intra_kernel<<<wblocks(N), blk, 0, stream>>>(nei_d0, h0, refp + 0 * N, neip + 0 * N, emb + (size_t)0 * N * HID, N);
  intra_kernel<<<wblocks(N), blk, 0, stream>>>(nei_d1, h1, refp + 1 * N, neip + 1 * N, emb + (size_t)1 * N * HID, N);
  intra_kernel<<<wblocks(N), blk, 0, stream>>>(nei_p0, h0, refp + 2 * N, neip + 2 * N, emb + (size_t)2 * N * HID, N);
  intra_kernel<<<wblocks(N), blk, 0, stream>>>(nei_p1, h1, refp + 3 * N, neip + 3 * N, emb + (size_t)3 * N * HID, N);
  intra_kernel<<<wblocks(N), blk, 0, stream>>>(nei_p2, h3, refp + 4 * N, neip + 4 * N, emb + (size_t)4 * N * HID, N);

  // --- semantic attention: colsum of tanh(E @ W^T + b) via WMMA
  hipMemsetAsync(colsum, 0, (size_t)5 * HID * sizeof(float), stream);
  int ntiles = N >> 4;                       // 3125 (N is a multiple of 16)
  dim3 fcgrid((unsigned)((ntiles + 7) / 8)); // one wave per tile
  fc_tanh_colsum_kernel<<<fcgrid, blk, 0, stream>>>(emb + (size_t)0 * N * HID, fc_w_d, fc_b_d, colsum + 0 * HID, N);
  fc_tanh_colsum_kernel<<<fcgrid, blk, 0, stream>>>(emb + (size_t)1 * N * HID, fc_w_d, fc_b_d, colsum + 1 * HID, N);
  fc_tanh_colsum_kernel<<<fcgrid, blk, 0, stream>>>(emb + (size_t)2 * N * HID, fc_w_p, fc_b_p, colsum + 2 * HID, N);
  fc_tanh_colsum_kernel<<<fcgrid, blk, 0, stream>>>(emb + (size_t)3 * N * HID, fc_w_p, fc_b_p, colsum + 3 * HID, N);
  fc_tanh_colsum_kernel<<<fcgrid, blk, 0, stream>>>(emb + (size_t)4 * N * HID, fc_w_p, fc_b_p, colsum + 4 * HID, N);

  beta_kernel<<<1, 32, 0, stream>>>(colsum + 0 * HID, sem_d, betas + 0, 2, 1.0f / (float)N);
  beta_kernel<<<1, 32, 0, stream>>>(colsum + 2 * HID, sem_p, betas + 2, 3, 1.0f / (float)N);

  // --- combine: z_d then z_p, concatenated in d_out
  int total = N * HID;
  dim3 cgrid((unsigned)((total + 255) / 256));
  combine_kernel<<<cgrid, blk, 0, stream>>>(emb + (size_t)0 * N * HID, emb + (size_t)1 * N * HID,
                                            emb + (size_t)0 * N * HID, betas + 0, 2,
                                            (float*)d_out, total);
  combine_kernel<<<cgrid, blk, 0, stream>>>(emb + (size_t)2 * N * HID, emb + (size_t)3 * N * HID,
                                            emb + (size_t)4 * N * HID, betas + 2, 3,
                                            (float*)d_out + total, total);
}
